// ActionLearner_83846351553003
// MI455X (gfx1250) — matrix operand order repaired and checked by reading, not compiled, not run on hardware
//
#include <hip/hip_runtime.h>
#include <hip/hip_bf16.h>
#include <math.h>

// Problem sizes (fixed by the reference)
#define BB   64
#define TT   256
#define IN_  512
#define HID_ 1024
#define OUT_ 256

#define KT   32            // K per WMMA step (bf16 16x16x32)
#define KP   40            // padded K stride (bf16 units): 80B rows, 16B aligned
#define MT   64            // T rows per workgroup
#define HP   (HID_ + 8)    // Hs row stride in bf16 (2064B, 16B aligned)

#define HS_USHORT   (MT * HP)              // 66,048 halfwords = 132,096 B
#define STAGE_BYTES (OUT_ * KP * 2)        // 20,480 B (phase-2 stage; >= phase-1 stage)
#define SMEM_BYTES  (HS_USHORT * 2 + STAGE_BYTES)   // 152,576 B < 320 KB

typedef __attribute__((ext_vector_type(16))) __bf16 v16bf;
typedef __attribute__((ext_vector_type(8)))  float  v8f;

// fp32 -> bf16, round-to-nearest-even (scalar fallback)
static __device__ __forceinline__ unsigned short f2bf(float f) {
    unsigned int u = __builtin_bit_cast(unsigned int, f);
    u += 0x7FFFu + ((u >> 16) & 1u);
    return (unsigned short)(u >> 16);
}

// Two fp32 -> packed bf16x2 (low = a, high = b).
static __device__ __forceinline__ unsigned int f2bf_pk(float a, float b) {
#if __has_builtin(__builtin_amdgcn_cvt_pk_bf16_f32)
    auto r = __builtin_amdgcn_cvt_pk_bf16_f32(a, b);
    return __builtin_bit_cast(unsigned int, r);
#else
    return (unsigned int)f2bf(a) | ((unsigned int)f2bf(b) << 16);
#endif
}

struct U128x2 { uint4 a, b; };

// Build a 16-halfword WMMA fragment from two 16-byte aligned LDS chunks.
static __device__ __forceinline__ v16bf ld2(const unsigned short* p0,
                                            const unsigned short* p1) {
    U128x2 t;
    t.a = *(const uint4*)p0;
    t.b = *(const uint4*)p1;
    return __builtin_bit_cast(v16bf, t);
}

// Exact-erf GELU via branchless Abramowitz-Stegun 7.1.26 (|err| <= 1.5e-7,
// i.e. exact at bf16 resolution). One v_rcp, one v_exp, no EXEC-mask branches.
static __device__ __forceinline__ float gelu_erf(float v) {
    float x  = v * 0.7071067811865475f;
    float ax = __builtin_fabsf(x);
    float t  = __builtin_amdgcn_rcpf(__builtin_fmaf(0.3275911f, ax, 1.0f));
    float p  = __builtin_fmaf(1.061405429f, t, -1.453152027f);
    p = __builtin_fmaf(p, t, 1.421413741f);
    p = __builtin_fmaf(p, t, -0.284496736f);
    p = __builtin_fmaf(p, t, 0.254829592f);
    p = p * t;
    float e  = __expf(-ax * ax);
    float r  = __builtin_fmaf(-p, e, 1.0f);          // erf(|x|)
    unsigned int ur = __builtin_bit_cast(unsigned int, r);
    unsigned int ux = __builtin_bit_cast(unsigned int, x);
    float rs = __builtin_bit_cast(float, (ur & 0x7fffffffu) | (ux & 0x80000000u));
    return 0.5f * v * (1.0f + rs);
}

__global__ void __launch_bounds__(256)
domain_mlp_fused(const float* __restrict__ x,
                 const int*   __restrict__ hetero,      // (B,2) first field = domain
                 const float* __restrict__ fc1_table,   // (20, IN*HID + HID)
                 const float* __restrict__ fc2_table,   // (20, HID*OUT + OUT)
                 float*       __restrict__ out) {
    extern __shared__ char smem[];
    unsigned short* Hs   = (unsigned short*)smem;                       // [MT][HP] bf16
    unsigned short* Asg  = (unsigned short*)(smem + HS_USHORT * 2);     // [MT][KP]   (phase 1)
    unsigned short* Bsg  = Asg + MT * KP;                               // [128][KP]  (phase 1)
    unsigned short* Bsg2 = (unsigned short*)(smem + HS_USHORT * 2);     // [256][KP]  (phase 2)

    const int tid  = threadIdx.x;
    const int lane = tid & 31;
    const int wid  = tid >> 5;          // 8 waves
    const int m16  = lane & 15;
    const int hsel = lane >> 4;

    const int b   = blockIdx.x >> 2;    // batch
    const int ts  = blockIdx.x & 3;     // T slice
    const int tm0 = ts * MT;

    const int dom = hetero[2 * b];
    const float* w1 = fc1_table + (size_t)dom * (IN_ * HID_ + HID_);
    const float* b1 = w1 + (size_t)IN_ * HID_;
    const float* w2 = fc2_table + (size_t)dom * (HID_ * OUT_ + OUT_);
    const float* b2 = w2 + (size_t)HID_ * OUT_;
    const float* xB = x   + ((size_t)b * TT + tm0) * IN_;
    float*      oB  = out + ((size_t)b * TT + tm0) * OUT_;

    // ---------------- Phase 1: H = gelu(X @ W1 + b1) -> LDS (bf16) ----------
    // Output tile per n-iter: 64 x 128. Waves: 2(M) x 4(N), each 32x32 (2x2 frags).
    {
        const int wm = wid & 1;         // 0..1 : 32-row half
        const int wn = wid >> 1;        // 0..3 : 32-col strip
        for (int nt = 0; nt < 8; ++nt) {
            const int n0 = nt * 128;
            v8f acc[2][2] = {};
            for (int kt = 0; kt < IN_ / KT; ++kt) {
                const int k0 = kt * KT;
                // Stage A: 64x32 fp32 -> bf16 [row][k], coalesced float4 loads.
#pragma unroll
                for (int p = 0; p < 2; ++p) {
                    const int idx = p * 256 + tid;
                    const int row = idx >> 3;
                    const int kq  = (idx & 7) * 4;
                    float4 v = *(const float4*)(xB + (size_t)row * IN_ + k0 + kq);
                    *(uint2*)(Asg + row * KP + kq) =
                        make_uint2(f2bf_pk(v.x, v.y), f2bf_pk(v.z, v.w));
                }
                // Stage B: W1 tile 32(K) x 128(N) fp32 -> transposed bf16 [n][k].
#pragma unroll
                for (int p = 0; p < 4; ++p) {
                    const int idx = p * 256 + tid;
                    const int k  = idx >> 5;
                    const int nq = (idx & 31) * 4;
                    float4 v = *(const float4*)(w1 + (size_t)(k0 + k) * HID_ + n0 + nq);
                    unsigned int r01 = f2bf_pk(v.x, v.y);
                    unsigned int r23 = f2bf_pk(v.z, v.w);
                    Bsg[(nq + 0) * KP + k] = (unsigned short)r01;
                    Bsg[(nq + 1) * KP + k] = (unsigned short)(r01 >> 16);
                    Bsg[(nq + 2) * KP + k] = (unsigned short)r23;
                    Bsg[(nq + 3) * KP + k] = (unsigned short)(r23 >> 16);
                }
                __syncthreads();

                v16bf af[2], bf[2];
#pragma unroll
                for (int fm = 0; fm < 2; ++fm) {
                    const unsigned short* pa = Asg + (wm * 32 + fm * 16 + m16) * KP + hsel * 8;
                    af[fm] = ld2(pa, pa + 16);
                }
#pragma unroll
                for (int fn = 0; fn < 2; ++fn) {
                    const unsigned short* pb = Bsg + (wn * 32 + fn * 16 + m16) * KP + hsel * 8;
                    bf[fn] = ld2(pb, pb + 16);
                }
#pragma unroll
                for (int fm = 0; fm < 2; ++fm)
#pragma unroll
                    for (int fn = 0; fn < 2; ++fn)
                        acc[fm][fn] = __builtin_amdgcn_wmma_f32_16x16x32_bf16(
                            false, af[fm], false, bf[fn], (short)0, acc[fm][fn],
                            false, false);
                __syncthreads();
            }
            // Epilogue: bias + GELU, packed bf16 conversion, write into Hs.
#pragma unroll
            for (int fm = 0; fm < 2; ++fm)
#pragma unroll
                for (int fn = 0; fn < 2; ++fn) {
                    const int col  = n0 + wn * 32 + fn * 16 + m16;
                    const float bv = b1[col];
                    const int rowb = wm * 32 + fm * 16 + hsel * 8;
#pragma unroll
                    for (int r = 0; r < 8; r += 2) {
                        float g0 = gelu_erf(acc[fm][fn][r]     + bv);
                        float g1 = gelu_erf(acc[fm][fn][r + 1] + bv);
                        unsigned int pk = f2bf_pk(g0, g1);
                        Hs[(rowb + r)     * HP + col] = (unsigned short)pk;
                        Hs[(rowb + r + 1) * HP + col] = (unsigned short)(pk >> 16);
                    }
                }
        }
    }
    __syncthreads();   // Hs complete; staging region is about to be reused.

    // ---------------- Phase 2: Out = H @ W2 + b2 ----------------------------
    // Output 64 x 256. Waves: 2(M) x 4(N), each 32x64 (2x4 frags).
    {
        const int wm = wid & 1;
        const int wn = wid >> 1;
        v8f acc[2][4] = {};
        for (int kt = 0; kt < HID_ / KT; ++kt) {
            const int k0 = kt * KT;
            // Stage B: W2 tile 32(K) x 256(N) fp32 -> transposed bf16 [n][k].
#pragma unroll
            for (int p = 0; p < 8; ++p) {
                const int idx = p * 256 + tid;
                const int k  = idx >> 6;
                const int nq = (idx & 63) * 4;
                float4 v = *(const float4*)(w2 + (size_t)(k0 + k) * OUT_ + nq);
                unsigned int r01 = f2bf_pk(v.x, v.y);
                unsigned int r23 = f2bf_pk(v.z, v.w);
                Bsg2[(nq + 0) * KP + k] = (unsigned short)r01;
                Bsg2[(nq + 1) * KP + k] = (unsigned short)(r01 >> 16);
                Bsg2[(nq + 2) * KP + k] = (unsigned short)r23;
                Bsg2[(nq + 3) * KP + k] = (unsigned short)(r23 >> 16);
            }
            __syncthreads();

            v16bf af[2], bf[4];
#pragma unroll
            for (int fm = 0; fm < 2; ++fm) {
                const unsigned short* pa = Hs + (wm * 32 + fm * 16 + m16) * HP + k0 + hsel * 8;
                af[fm] = ld2(pa, pa + 16);
            }
#pragma unroll
            for (int fn = 0; fn < 4; ++fn) {
                const unsigned short* pb = Bsg2 + (wn * 64 + fn * 16 + m16) * KP + hsel * 8;
                bf[fn] = ld2(pb, pb + 16);
            }
#pragma unroll
            for (int fm = 0; fm < 2; ++fm)
#pragma unroll
                for (int fn = 0; fn < 4; ++fn)
                    acc[fm][fn] = __builtin_amdgcn_wmma_f32_16x16x32_bf16(
                        false, af[fm], false, bf[fn], (short)0, acc[fm][fn],
                        false, false);
            __syncthreads();
        }
        // Epilogue: bias + fp32 store.
#pragma unroll
        for (int fm = 0; fm < 2; ++fm)
#pragma unroll
            for (int fn = 0; fn < 4; ++fn) {
                const int col  = wn * 64 + fn * 16 + m16;
                const float bv = b2[col];
#pragma unroll
                for (int r = 0; r < 8; ++r) {
                    const int row = wm * 32 + fm * 16 + hsel * 8 + r;
                    oB[(size_t)row * OUT_ + col] = acc[fm][fn][r] + bv;
                }
            }
    }
}

extern "C" void kernel_launch(void* const* d_in, const int* in_sizes, int n_in,
                              void* d_out, int out_size, void* d_ws, size_t ws_size,
                              hipStream_t stream) {
    (void)in_sizes; (void)n_in; (void)d_ws; (void)ws_size; (void)out_size;
    const float* x          = (const float*)d_in[0];
    const int*   hetero     = (const int*)d_in[1];
    const float* fc1_table  = (const float*)d_in[2];
    const float* fc2_table  = (const float*)d_in[3];
    float*       out        = (float*)d_out;

    // 149 KB dynamic LDS per workgroup (CDNA5 allows up to 320 KB/WGP);
    // raise the per-kernel limit (idempotent, not a stream op -> capture safe).
    hipFuncSetAttribute((const void*)domain_mlp_fused,
                        hipFuncAttributeMaxDynamicSharedMemorySize, SMEM_BYTES);

    dim3 grid(BB * (TT / MT));   // 64 batches x 4 T-slices = 256 WGs
    dim3 block(256);             // 8 wave32
    domain_mlp_fused<<<grid, block, SMEM_BYTES, stream>>>(x, hetero, fc1_table,
                                                          fc2_table, out);
}
